// LPDNetOrign_6262062318058
// MI455X (gfx1250) — matrix operand order repaired and checked by reading, not compiled, not run on hardware
//
#include <hip/hip_runtime.h>

// ---------------------------------------------------------------------------
// LPDNet (DGCNN) forward for MI455X / gfx1250.
// All channel matmuls via v_wmma_f32_16x16x32_bf16 (wave32 WMMA).
// Activations kept POINT-MAJOR bf16 [n][c] so WMMA A/B fragments are
// contiguous dword loads and epilogues are packed b128 stores.
// EdgeConv blocks fused (gather -> conv -> conv -> max over K): the
// (B,128,N,20) graph features (335 MB) never touch HBM.
// ---------------------------------------------------------------------------

#define BATCH 8
#define NPTS  4096
#define KNN   20

typedef __attribute__((ext_vector_type(16))) __bf16   v16bf;
typedef __attribute__((ext_vector_type(8)))  float    v8f;
typedef __attribute__((ext_vector_type(4)))  unsigned v4u;

union FragBF { v16bf v; unsigned u[8]; __bf16 e[16]; };
union OutPack { __bf16 e[8]; v4u q; };

__device__ static inline v8f vzero8() {
  v8f z = {0.f, 0.f, 0.f, 0.f, 0.f, 0.f, 0.f, 0.f};
  return z;
}

__device__ static inline v8f wmma_bf16(v16bf a, v16bf b, v8f c) {
  // D(16x16,f32) = A(16x32,bf16) x B(32x16,bf16) + C
  return __builtin_amdgcn_wmma_f32_16x16x32_bf16(
      /*neg_a=*/false, a, /*neg_b=*/false, b,
      /*c_mod=*/(short)0, c, /*reuse_a=*/false, /*reuse_b=*/false);
}

// A fragment (16x32 bf16) from row-major M[m][k] (k contiguous) -> dword loads.
// ISA layout: lanes 0-15 m=lane, v0..3 K=k0+0..7, v4..7 K=k0+16..23;
// lanes 16-31 same m, K bases +8 / +24.
__device__ static inline v16bf load_a_frag(const __bf16* W, int ld, int m0, int k0) {
  int lane = threadIdx.x & 31;
  const __bf16* row = W + (size_t)(m0 + (lane & 15)) * ld + k0 + ((lane >> 4) << 3);
  const unsigned* p0 = (const unsigned*)row;
  const unsigned* p1 = (const unsigned*)(row + 16);
  FragBF f;
  f.u[0] = p0[0]; f.u[1] = p0[1]; f.u[2] = p0[2]; f.u[3] = p0[3];
  f.u[4] = p1[0]; f.u[5] = p1[1]; f.u[6] = p1[2]; f.u[7] = p1[3];
  return f.v;
}

// B fragment (32x16 bf16) from column-major-per-point tile X[col][k], stride ld
// (even). Same K layout as A: lanes 0-15 (col = c0+lane) v0..3 K=k0+0..7,
// v4..7 K=k0+16..23; hi lanes K bases +8 / +24.
// Works for both global (point-major activations) and LDS tiles.
__device__ static inline v16bf load_b_frag(const __bf16* X, int ld, int c0, int k0) {
  int lane = threadIdx.x & 31;
  const __bf16* col = X + (size_t)(c0 + (lane & 15)) * ld + k0 + ((lane >> 4) << 3);
  const unsigned* p0 = (const unsigned*)col;
  const unsigned* p1 = (const unsigned*)(col + 16);
  FragBF f;
#pragma unroll
  for (int i = 0; i < 4; i++) { f.u[i] = p0[i]; f.u[i + 4] = p1[i]; }
  return f.v;
}

// Order-preserving float <-> u32 mapping for ds_max_u32-based float max.
__device__ static inline unsigned fenc(float f) {
  unsigned u = __float_as_uint(f);
  return (u & 0x80000000u) ? ~u : (u | 0x80000000u);
}
__device__ static inline float fdec(unsigned u) {
  u = (u & 0x80000000u) ? (u & 0x7fffffffu) : ~u;
  return __uint_as_float(u);
}

// Fully unrolled insertion into a sorted (descending) top-K list in registers.
__device__ static inline void top_insert(float v, int id, float* tv, int* ti) {
  if (v <= tv[KNN - 1]) return;
  tv[KNN - 1] = v; ti[KNN - 1] = id;
#pragma unroll
  for (int s = KNN - 1; s > 0; --s) {
    if (tv[s] > tv[s - 1]) {
      float tf = tv[s]; tv[s] = tv[s - 1]; tv[s - 1] = tf;
      int   t2 = ti[s]; ti[s] = ti[s - 1]; ti[s - 1] = t2;
    }
  }
}

// ---------------------------------------------------------------------------
// Prep kernels
// ---------------------------------------------------------------------------
__global__ void k_prep_bn(const float* __restrict__ bn, float* __restrict__ sc,
                          float* __restrict__ sh, int C) {
  int c = blockIdx.x * blockDim.x + threadIdx.x;
  if (c < C) {
    float g = bn[c], b = bn[C + c], m = bn[2 * C + c], v = bn[3 * C + c];
    float s = g * rsqrtf(v + 1e-5f);
    sc[c] = s;
    sh[c] = b - m * s;
  }
}

__global__ void k_w2bf(const float* __restrict__ w, __bf16* __restrict__ o, int n) {
  int i = blockIdx.x * blockDim.x + threadIdx.x;
  if (i < n) o[i] = (__bf16)w[i];
}

// conv1 (K=3, VALU): one thread per point computes all 64 channels and writes
// one contiguous 128B row (8 x b128 stores). x:(B,1,N,3) flat = (B*N,3).
__global__ void k_conv1(const float* __restrict__ x, const float* __restrict__ w1,
                        const float* __restrict__ sc, const float* __restrict__ sh,
                        __bf16* __restrict__ h1t) {
  int t = blockIdx.x * blockDim.x + threadIdx.x;   // over B*N points
  if (t >= BATCH * NPTS) return;
  const float* xp = x + (size_t)t * 3;
  float x0 = xp[0], x1 = xp[1], x2 = xp[2];
  __bf16 o[64];
#pragma unroll
  for (int co = 0; co < 64; co++) {
    float a = w1[co * 3] * x0 + w1[co * 3 + 1] * x1 + w1[co * 3 + 2] * x2;
    a = a * sc[co] + sh[co];
    a = a > 0.f ? a : 0.01f * a;
    o[co] = (__bf16)a;
  }
  v4u* dst = (v4u*)(h1t + (size_t)t * 64);
  const v4u* src = (const v4u*)o;
#pragma unroll
  for (int i = 0; i < 8; i++) dst[i] = src[i];
}

// Squared norms of the 64-d features (rows are contiguous in point-major).
__global__ void k_xx(const __bf16* __restrict__ Ht, float* __restrict__ xx) {
  int t = blockIdx.x * blockDim.x + threadIdx.x;
  if (t >= BATCH * NPTS) return;
  const __bf16* hp = Ht + (size_t)t * 64;
  float s = 0.f;
#pragma unroll 16
  for (int c = 0; c < 64; c++) {
    float v = (float)hp[c];
    s += v * v;
  }
  xx[t] = s;
}

// ---------------------------------------------------------------------------
// WMMA GEMM + BN + LeakyReLU on point-major activations. LDS-free:
// each wave owns one 16-point tile (B fragments resident in VGPRs) and sweeps
// 4 M-tiles. Epilogue: 8 consecutive channels per lane -> one b128 store.
// Grid (N/64, CO/64, B), block 128 (4 waves).
// ---------------------------------------------------------------------------
template <int CI, int CO>
__global__ void k_gemm_bn_act(const __bf16* __restrict__ W,
                              const float* __restrict__ sc, const float* __restrict__ sh,
                              const __bf16* __restrict__ Xt,
                              __bf16* __restrict__ Ybt, float* __restrict__ Yfc) {
  constexpr int KS = CI / 32;
  int b = blockIdx.z;
  int n0 = blockIdx.x * 64;
  int mbase = blockIdx.y * 64;
  int tid = threadIdx.x;
  int wave = tid >> 5, lane = tid & 31;
  const __bf16* Xb = Xt + (size_t)b * NPTS * CI;

  int nt = wave;                       // wave owns one 16-point tile
  v16bf bf[KS];
#pragma unroll
  for (int ks = 0; ks < KS; ks++) bf[ks] = load_b_frag(Xb, CI, n0 + nt * 16, ks * 32);

  if (n0 + 64 < NPTS)                  // pull next point-block toward the caches
    __builtin_prefetch(Xb + (size_t)(n0 + 64 + nt * 16 + (lane & 15)) * CI, 0, 1);

  int mhi = (lane >> 4) << 3;
  int n = n0 + nt * 16 + (lane & 15);
#pragma unroll
  for (int mt = 0; mt < 4; mt++) {
    int m0 = mbase + mt * 16;
    v8f acc = vzero8();
#pragma unroll
    for (int ks = 0; ks < KS; ks++)
      acc = wmma_bf16(load_a_frag(W, CI, m0, ks * 32), bf[ks], acc);

    int cobase = m0 + mhi;             // 8 consecutive channels for this lane
    OutPack ob;
#pragma unroll
    for (int r = 0; r < 8; r++) {
      float y = acc[r] * sc[cobase + r] + sh[cobase + r];
      y = y > 0.f ? y : 0.01f * y;
      ob.e[r] = (__bf16)y;
      if (Yfc) Yfc[((size_t)b * CO + cobase + r) * NPTS + n] = y;  // required output layout
    }
    if (Ybt) *(v4u*)(Ybt + ((size_t)b * NPTS + n) * CO + cobase) = ob.q;
  }
}

// ---------------------------------------------------------------------------
// Feature kNN: S = H H^T via WMMA on point-major H (all fragments are dword
// loads), chunked neg-distance rows in LDS, register insertion top-20.
// Grid (N/16, B), block 128 (4 waves).
// ---------------------------------------------------------------------------
__global__ void k_knn_feat(const __bf16* __restrict__ Ht, const float* __restrict__ xx,
                           int* __restrict__ idx) {
  int b = blockIdx.y;
  int q0 = blockIdx.x * 16;
  int tid = threadIdx.x;
  int wave = tid >> 5, lane = tid & 31;
  const __bf16* Hb = Ht + (size_t)b * NPTS * 64;
  const float* xb = xx + (size_t)b * NPTS;

  __shared__ float Sch[16][128];

  v16bf a0 = load_a_frag(Hb, 64, q0, 0);   // A = 16 query rows (k contiguous)
  v16bf a1 = load_a_frag(Hb, 64, q0, 32);

  int mhi = (lane >> 4) << 3;
  float xq[8];
#pragma unroll
  for (int r = 0; r < 8; r++) xq[r] = xb[q0 + r + mhi];

  float tv[KNN]; int ti[KNN];
#pragma unroll
  for (int i = 0; i < KNN; i++) { tv[i] = -3.4e38f; ti[i] = 0; }

  for (int c0 = 0; c0 < NPTS; c0 += 128) {
    if (c0 + 128 < NPTS)
      __builtin_prefetch(Hb + (size_t)(c0 + 128 + (lane & 15)) * 64, 0, 1);
    for (int t = wave; t < 8; t += 4) {      // uniform per wave; EXEC stays full
      int p0 = c0 + t * 16;
      v8f acc = vzero8();
      acc = wmma_bf16(a0, load_b_frag(Hb, 64, p0, 0), acc);
      acc = wmma_bf16(a1, load_b_frag(Hb, 64, p0, 32), acc);
      float xp = xb[p0 + (lane & 15)];
#pragma unroll
      for (int r = 0; r < 8; r++)
        Sch[r + mhi][t * 16 + (lane & 15)] = 2.f * acc[r] - xq[r] - xp;
    }
    __syncthreads();
    if (tid < 16) {
      for (int j = 0; j < 128; j++) top_insert(Sch[tid][j], c0 + j, tv, ti);
    }
    __syncthreads();
  }
  if (tid < 16) {
    int q = q0 + tid;
#pragma unroll
    for (int kk = 0; kk < KNN; kk++)
      idx[((size_t)b * NPTS + q) * KNN + kk] = ti[kk];
  }
}

// Coordinate kNN (3-d): VALU with LDS candidate staging. Grid (N/128, B).
__global__ void k_knn3(const float* __restrict__ x, int* __restrict__ idx) {
  int b = blockIdx.y;
  int tid = threadIdx.x;
  int q = blockIdx.x * 128 + tid;
  const float* xb = x + (size_t)b * NPTS * 3;
  float qx = xb[q * 3], qy = xb[q * 3 + 1], qz = xb[q * 3 + 2];
  __shared__ float cs[128 * 3];
  float tv[KNN]; int ti[KNN];
#pragma unroll
  for (int i = 0; i < KNN; i++) { tv[i] = -3.4e38f; ti[i] = 0; }

  for (int j0 = 0; j0 < NPTS; j0 += 128) {
    __syncthreads();
    for (int e = tid; e < 384; e += 128) cs[e] = xb[(size_t)j0 * 3 + e];
    __syncthreads();
    for (int jj = 0; jj < 128; jj++) {
      float dx = qx - cs[jj * 3], dy = qy - cs[jj * 3 + 1], dz = qz - cs[jj * 3 + 2];
      top_insert(-(dx * dx + dy * dy + dz * dz), j0 + jj, tv, ti);
    }
  }
#pragma unroll
  for (int kk = 0; kk < KNN; kk++)
    idx[((size_t)b * NPTS + q) * KNN + kk] = ti[kk];
}

// ---------------------------------------------------------------------------
// Fused EdgeConv: gather(+center diff) -> WMMA conv1 -> act -> WMMA conv2 ->
// act -> max over K=20 (LDS ds_max_u32). 8 points / block, 256 threads.
// CATMODE=1: CI=128 ([h[nbr]-h[n]; h[n]]); CATMODE=0: CI=64 (gather only).
// Input Ht is point-major (N,64) bf16 -> gathers are contiguous rows.
// ---------------------------------------------------------------------------
template <int CATMODE>
__global__ void k_edgeconv(const __bf16* __restrict__ Hin, const int* __restrict__ idx,
                           const __bf16* __restrict__ W1, const float* __restrict__ sc1,
                           const float* __restrict__ sh1,
                           const __bf16* __restrict__ W2, const float* __restrict__ sc2,
                           const float* __restrict__ sh2,
                           __bf16* __restrict__ Out) {
  constexpr int CI   = CATMODE ? 128 : 64;
  constexpr int CSH  = CATMODE ? 7 : 6;
  constexpr int LST1 = CI + 2;
  constexpr int LST2 = 66;
  constexpr int NCOL = 8 * KNN;  // 160 columns = 8 points x 20 neighbors
  constexpr size_t GB = (size_t)NCOL * LST1 * sizeof(__bf16);
  constexpr size_t ZB = (size_t)NCOL * LST2 * sizeof(__bf16);
  __shared__ __align__(16) char smem[GB + ZB];   // <= 62.7 KB
  __shared__ int nbrs[NCOL];
  __bf16* G = (__bf16*)smem;
  __bf16* Z = (__bf16*)(smem + GB);
  unsigned* OM = (unsigned*)smem;                // overlays G after GEMM1 is done

  int b = blockIdx.y;
  int n0 = blockIdx.x * 8;
  int tid = threadIdx.x;
  const __bf16* Hb = Hin + (size_t)b * NPTS * 64;

  if (tid < NCOL) {
    int p = tid / KNN, k = tid - p * KNN;
    nbrs[tid] = idx[((size_t)b * NPTS + n0 + p) * KNN + k];
  }
  __syncthreads();

  // Stage graph-feature columns into LDS [col][ci]; consecutive threads read
  // consecutive channels of a gathered row (coalesced).
  for (int e = tid; e < NCOL * CI; e += 256) {
    int ci  = e & (CI - 1);
    int col = e >> CSH;
    int n = n0 + (col / KNN);
    int nbr = nbrs[col];
    float v;
    if (CATMODE) {
      v = (ci < 64) ? ((float)Hb[(size_t)nbr * 64 + ci] - (float)Hb[(size_t)n * 64 + ci])
                    : (float)Hb[(size_t)n * 64 + (ci - 64)];
    } else {
      v = (float)Hb[(size_t)nbr * 64 + ci];
    }
    G[(size_t)col * LST1 + ci] = (__bf16)v;
  }
  __syncthreads();

  int wave = tid >> 5, lane = tid & 31;
  int mt = wave & 3;       // 4 M-tiles cover CO=64
  int cg = wave >> 2;      // 2 wave-groups over 10 column tiles
  int mhi = (lane >> 4) << 3;
  constexpr int KS1 = CI / 32;

  v16bf a1f[KS1];
#pragma unroll
  for (int ks = 0; ks < KS1; ks++) a1f[ks] = load_a_frag(W1, CI, mt * 16, ks * 32);

  for (int ct = cg; ct < NCOL / 16; ct += 2) {
    v8f acc = vzero8();
#pragma unroll
    for (int ks = 0; ks < KS1; ks++)
      acc = wmma_bf16(a1f[ks], load_b_frag(G, LST1, ct * 16, ks * 32), acc);
    int coln = ct * 16 + (lane & 15);
#pragma unroll
    for (int r = 0; r < 8; r++) {
      int co = mt * 16 + r + mhi;
      float y = acc[r] * sc1[co] + sh1[co];
      y = y > 0.f ? y : 0.01f * y;
      Z[(size_t)coln * LST2 + co] = (__bf16)y;
    }
  }
  __syncthreads();                 // GEMM1 done: G dead, Z complete
  for (int e = tid; e < 64 * 8; e += 256) OM[e] = 0u;   // enc(-NaN) == minimum
  __syncthreads();

  v16bf a2f[2];
  a2f[0] = load_a_frag(W2, 64, mt * 16, 0);
  a2f[1] = load_a_frag(W2, 64, mt * 16, 32);
  for (int ct = cg; ct < NCOL / 16; ct += 2) {
    v8f acc = vzero8();
    acc = wmma_bf16(a2f[0], load_b_frag(Z, LST2, ct * 16, 0), acc);
    acc = wmma_bf16(a2f[1], load_b_frag(Z, LST2, ct * 16, 32), acc);
    int coln = ct * 16 + (lane & 15);
    int p = coln / KNN;
#pragma unroll
    for (int r = 0; r < 8; r++) {
      int co = mt * 16 + r + mhi;
      float y = acc[r] * sc2[co] + sh2[co];
      y = y > 0.f ? y : 0.01f * y;
      atomicMax(&OM[co * 8 + p], fenc(y));   // ds_max_u32
    }
  }
  __syncthreads();
  // Point-major output: consecutive threads -> consecutive channels.
  for (int e = tid; e < 64 * 8; e += 256) {
    int co = e & 63, p = e >> 6;
    Out[((size_t)b * NPTS + n0 + p) * 64 + co] = (__bf16)fdec(OM[co * 8 + p]);
  }
}

// ---------------------------------------------------------------------------
extern "C" void kernel_launch(void* const* d_in, const int* in_sizes, int n_in,
                              void* d_out, int out_size, void* d_ws, size_t ws_size,
                              hipStream_t stream) {
  (void)in_sizes; (void)n_in; (void)out_size; (void)ws_size;
  const float* x     = (const float*)d_in[0];
  const float* w1    = (const float*)d_in[1];
  const float* bn1   = (const float*)d_in[2];
  const float* w2    = (const float*)d_in[3];
  const float* bn2   = (const float*)d_in[4];
  const float* wdg1  = (const float*)d_in[5];
  const float* bndg1 = (const float*)d_in[6];
  const float* wdg2  = (const float*)d_in[7];
  const float* bndg2 = (const float*)d_in[8];
  const float* wsn1  = (const float*)d_in[9];
  const float* bnsn1 = (const float*)d_in[10];
  const float* wsn2  = (const float*)d_in[11];
  const float* bnsn2 = (const float*)d_in[12];
  const float* w3    = (const float*)d_in[13];
  const float* bn3   = (const float*)d_in[14];
  const float* w4    = (const float*)d_in[15];
  const float* bn4   = (const float*)d_in[16];
  const float* w5    = (const float*)d_in[17];
  const float* bn5   = (const float*)d_in[18];

  char* ws = (char*)d_ws;
  size_t off = 0;
  auto alloc = [&](size_t bytes) -> void* {
    void* p = ws + off;
    off = (off + bytes + 255) & ~(size_t)255;
    return p;
  };

  // Point-major bf16 activations [b][n][c]
  __bf16* h1t = (__bf16*)alloc((size_t)BATCH * NPTS * 64 * 2);
  __bf16* h2t = (__bf16*)alloc((size_t)BATCH * NPTS * 64 * 2);
  __bf16* m1t = (__bf16*)alloc((size_t)BATCH * NPTS * 64 * 2);
  __bf16* m2t = (__bf16*)alloc((size_t)BATCH * NPTS * 64 * 2);
  __bf16* h3t = (__bf16*)alloc((size_t)BATCH * NPTS * 64 * 2);
  __bf16* h4t = (__bf16*)alloc((size_t)BATCH * NPTS * 128 * 2);
  float*  xxb = (float*)alloc((size_t)BATCH * NPTS * 4);
  int*   idx1 = (int*)alloc((size_t)BATCH * NPTS * KNN * 4);
  int*   idx2 = (int*)alloc((size_t)BATCH * NPTS * KNN * 4);

  __bf16* w2b   = (__bf16*)alloc(64 * 64 * 2);
  __bf16* wdg1b = (__bf16*)alloc(64 * 128 * 2);
  __bf16* wdg2b = (__bf16*)alloc(64 * 64 * 2);
  __bf16* wsn1b = (__bf16*)alloc(64 * 64 * 2);
  __bf16* wsn2b = (__bf16*)alloc(64 * 64 * 2);
  __bf16* w3b   = (__bf16*)alloc(64 * 64 * 2);
  __bf16* w4b   = (__bf16*)alloc(128 * 64 * 2);
  __bf16* w5b   = (__bf16*)alloc(512 * 128 * 2);

  float *sc1 = (float*)alloc(64 * 4),  *sh1 = (float*)alloc(64 * 4);
  float *sc2 = (float*)alloc(64 * 4),  *sh2 = (float*)alloc(64 * 4);
  float *scd1 = (float*)alloc(64 * 4), *shd1 = (float*)alloc(64 * 4);
  float *scd2 = (float*)alloc(64 * 4), *shd2 = (float*)alloc(64 * 4);
  float *scs1 = (float*)alloc(64 * 4), *shs1 = (float*)alloc(64 * 4);
  float *scs2 = (float*)alloc(64 * 4), *shs2 = (float*)alloc(64 * 4);
  float *sc3 = (float*)alloc(64 * 4),  *sh3 = (float*)alloc(64 * 4);
  float *sc4 = (float*)alloc(128 * 4), *sh4 = (float*)alloc(128 * 4);
  float *sc5 = (float*)alloc(512 * 4), *sh5 = (float*)alloc(512 * 4);

  auto bnprep = [&](const float* bn, float* sc, float* sh, int C) {
    k_prep_bn<<<dim3((C + 63) / 64), dim3(64), 0, stream>>>(bn, sc, sh, C);
  };
  bnprep(bn1, sc1, sh1, 64);    bnprep(bn2, sc2, sh2, 64);
  bnprep(bndg1, scd1, shd1, 64); bnprep(bndg2, scd2, shd2, 64);
  bnprep(bnsn1, scs1, shs1, 64); bnprep(bnsn2, scs2, shs2, 64);
  bnprep(bn3, sc3, sh3, 64);    bnprep(bn4, sc4, sh4, 128);
  bnprep(bn5, sc5, sh5, 512);

  auto wconv = [&](const float* w, __bf16* o, int n) {
    k_w2bf<<<dim3((n + 255) / 256), dim3(256), 0, stream>>>(w, o, n);
  };
  wconv(w2, w2b, 64 * 64);     wconv(wdg1, wdg1b, 64 * 128);
  wconv(wdg2, wdg2b, 64 * 64); wconv(wsn1, wsn1b, 64 * 64);
  wconv(wsn2, wsn2b, 64 * 64); wconv(w3, w3b, 64 * 64);
  wconv(w4, w4b, 128 * 64);    wconv(w5, w5b, 512 * 128);

  // conv1 (VALU, K=3) then conv2 (WMMA)
  k_conv1<<<dim3(BATCH * NPTS / 128), dim3(128), 0, stream>>>(x, w1, sc1, sh1, h1t);
  k_gemm_bn_act<64, 64><<<dim3(NPTS / 64, 1, BATCH), dim3(128), 0, stream>>>(
      w2b, sc2, sh2, h1t, h2t, nullptr);

  // kNN on learned features (WMMA Gram matrix) and on raw coords (VALU)
  k_xx<<<dim3(BATCH * NPTS / 256), dim3(256), 0, stream>>>(h2t, xxb);
  k_knn_feat<<<dim3(NPTS / 16, BATCH), dim3(128), 0, stream>>>(h2t, xxb, idx1);
  k_knn3<<<dim3(NPTS / 128, BATCH), dim3(128), 0, stream>>>(x, idx2);

  // Fused EdgeConv blocks
  k_edgeconv<1><<<dim3(NPTS / 8, BATCH), dim3(256), 0, stream>>>(
      h2t, idx1, wdg1b, scd1, shd1, wdg2b, scd2, shd2, m1t);
  k_edgeconv<0><<<dim3(NPTS / 8, BATCH), dim3(256), 0, stream>>>(
      m1t, idx2, wsn1b, scs1, shs1, wsn2b, scs2, shs2, m2t);

  // Final MLP 64 -> 64 -> 128 -> 512 (last layer writes fp32 d_out, c-major)
  k_gemm_bn_act<64, 64><<<dim3(NPTS / 64, 1, BATCH), dim3(128), 0, stream>>>(
      w3b, sc3, sh3, m2t, h3t, nullptr);
  k_gemm_bn_act<64, 128><<<dim3(NPTS / 64, 2, BATCH), dim3(128), 0, stream>>>(
      w4b, sc4, sh4, h3t, h4t, nullptr);
  k_gemm_bn_act<128, 512><<<dim3(NPTS / 64, 8, BATCH), dim3(128), 0, stream>>>(
      w5b, sc5, sh5, h4t, nullptr, (float*)d_out);
}
